// TopologyBRepEncoder_90752658964773
// MI455X (gfx1250) — compile-verified, matrix operand order checked by reading, not run on hardware
//
#include <hip/hip_runtime.h>
#include <hip/hip_bf16.h>
#include <math.h>

// ---------------------------------------------------------------------------
// TopologyBRepEncoder for MI455X (gfx1250, wave32, WMMA).
// Everything is L2-resident (192MB L2 vs ~45MB of weights+activations), so
// GEMMs feed v_wmma_f32_16x16x32_f16 with direct per-lane 32B fragment loads
// from f16 operands pre-laid-out in memory (weights transposed to [N][K] f16
// once per launch; activation producers emit f16 mirrors).  No LDS staging,
// no barriers in the GEMM hot loop.  Attention keeps f16 score rows in the
// 320KB WGP LDS and uses WMMA for both QK^T and PV.
// ---------------------------------------------------------------------------

typedef __attribute__((ext_vector_type(16))) _Float16 v16h;
typedef __attribute__((ext_vector_type(8)))  float    v8f;

#define B_      2
#define NF_     512
#define NE_     1024
#define D_      384
#define H_      8
#define HD_     48
#define HDP_    64          // head dim padded to 2 x K32 WMMA chunks
#define S_      (NF_ + NE_) // 1536
#define MAXLVL_ 32

__device__ __forceinline__ float gelu_f(float x) {
  return 0.5f * x * (1.0f + erff(x * 0.70710678118654752440f));
}

__device__ __forceinline__ float block_sum(float v, float* sbuf) {
  int t = threadIdx.x;
  sbuf[t] = v;
  __syncthreads();
  for (int off = 64; off > 0; off >>= 1) {
    if (t < off) sbuf[t] += sbuf[t + off];
    __syncthreads();
  }
  float r = sbuf[0];
  __syncthreads();
  return r;
}

// ---------------------------------------------------------------------------
// WMMA GEMM, all-f16 operands, no LDS:
//   C[M,N] = act(A16[M,K] @ Wt16[N,K]^T + bias) (+ res)
// Block = 128 thr = 4 waves; block tile 64x128, wave tile 32x64 (2x4 WMMA).
// A16 row-major [M][K], Wt16 row-major [N][K]: both fragment loads are
// contiguous 32B per lane.  ACT: 0 none, 1 gelu(erf), 2 sigmoid.
// Outputs: C (f32, nullable) and/or C16 (f16 mirror, nullable).
// ---------------------------------------------------------------------------
template<int ACT>
__global__ __launch_bounds__(128)
void gemm16_k(const _Float16* __restrict__ A, const _Float16* __restrict__ Wt,
              const float* __restrict__ bias, const float* __restrict__ res,
              float* __restrict__ C, _Float16* __restrict__ C16,
              int M, int N, int K) {
  const int t = threadIdx.x;
  const int lane = t & 31;
  const int wave = t >> 5;
  const int lr = lane & 15, lh = lane >> 4;
  const int m0 = blockIdx.y * 64 + (wave >> 1) * 32;
  const int n0 = blockIdx.x * 128 + (wave & 1) * 64;

  v8f acc[8];
  for (int i = 0; i < 8; ++i)
    for (int j = 0; j < 8; ++j) acc[i][j] = 0.0f;

  const _Float16* Ap0 = A + (size_t)(m0 + lr) * K + lh * 16;
  const _Float16* Ap1 = A + (size_t)(m0 + 16 + lr) * K + lh * 16;
  const _Float16* Bp[4];
#pragma unroll
  for (int ct = 0; ct < 4; ++ct)
    Bp[ct] = Wt + (size_t)(n0 + ct * 16 + lr) * K + lh * 16;

  for (int kt = 0; kt < K; kt += 32) {
    __builtin_prefetch(Ap0 + kt + 128, 0, 1);
    __builtin_prefetch(Bp[0] + kt + 128, 0, 1);
    v16h a0 = *(const v16h*)(Ap0 + kt);
    v16h a1 = *(const v16h*)(Ap1 + kt);
#pragma unroll
    for (int ct = 0; ct < 4; ++ct) {
      v16h bv = *(const v16h*)(Bp[ct] + kt);
      acc[ct]     = __builtin_amdgcn_wmma_f32_16x16x32_f16(false, a0, false, bv, (short)0, acc[ct],     false, false);
      acc[4 + ct] = __builtin_amdgcn_wmma_f32_16x16x32_f16(false, a1, false, bv, (short)0, acc[4 + ct], false, false);
    }
  }
#pragma unroll
  for (int rt = 0; rt < 2; ++rt) {
#pragma unroll
    for (int ct = 0; ct < 4; ++ct) {
      int mt = m0 + rt * 16 + ((lane < 16) ? 0 : 8);
      int n  = n0 + ct * 16 + lr;
      float bi = bias ? bias[n] : 0.0f;
      for (int r = 0; r < 8; ++r) {
        float v = acc[rt * 4 + ct][r] + bi;
        if (ACT == 1) v = gelu_f(v);
        else if (ACT == 2) v = 1.0f / (1.0f + __expf(-v));
        size_t idx = (size_t)(mt + r) * N + n;
        if (res) v += res[idx];
        if (C)   C[idx] = v;
        if (C16) C16[idx] = (_Float16)v;
      }
    }
  }
}

// Weight transpose+convert: Wt16[n][k] = (f16)W[k][n]   (runs every launch)
__global__ void wT16_k(const float* __restrict__ W, _Float16* __restrict__ Wt,
                       int K, int N) {
  int i = blockIdx.x * 256 + threadIdx.x;
  if (i >= K * N) return;
  int n = i % N, k = i / N;
  Wt[(size_t)n * K + k] = (_Float16)W[i];
}

// ---------------------------------------------------------------------------
// Attention: one block (1 wave) per (b, h, 16-query tile).
// Scores for all S keys held f16 in LDS; WMMA for QK^T and PV.
// Output written as f16 mirror (it only feeds the out-proj GEMM).
// ---------------------------------------------------------------------------
__global__ __launch_bounds__(32)
void attn_k(const _Float16* __restrict__ Qh, const _Float16* __restrict__ Kh,
            const _Float16* __restrict__ Vt, const float* __restrict__ KB,
            _Float16* __restrict__ Oout) {
  __shared__ __attribute__((aligned(32))) _Float16 P[16][S_];  // 48 KB
  __shared__ float invs[16];
  const int lane = threadIdx.x;
  const int lr = lane & 15, lh = lane >> 4;
  const int qt = blockIdx.x, h = blockIdx.y, b = blockIdx.z;
  const int qBase = qt * 16;
  const size_t hb = ((size_t)(b * H_ + h)) * S_;
  const float scale = 0.14433756729740643f;  // 1/sqrt(48)

  const _Float16* Qp = Qh + (hb + qBase) * HDP_;
  v16h aq0 = *(const v16h*)(Qp + lr * HDP_ +  0 + lh * 16);
  v16h aq1 = *(const v16h*)(Qp + lr * HDP_ + 32 + lh * 16);

  for (int kb = 0; kb < S_ / 16; ++kb) {
    const int keyBase = kb * 16;
    const _Float16* Kp = Kh + (hb + keyBase) * HDP_;
    v16h bk0 = *(const v16h*)(Kp + lr * HDP_ +  0 + lh * 16);
    v16h bk1 = *(const v16h*)(Kp + lr * HDP_ + 32 + lh * 16);
    v8f acc;
    for (int j = 0; j < 8; ++j) acc[j] = 0.0f;
    acc = __builtin_amdgcn_wmma_f32_16x16x32_f16(false, aq0, false, bk0, (short)0, acc, false, false);
    acc = __builtin_amdgcn_wmma_f32_16x16x32_f16(false, aq1, false, bk1, (short)0, acc, false, false);
    float kbv = KB[b * S_ + keyBase + lr];
    for (int r = 0; r < 8; ++r) {
      int m = (lane < 16) ? r : r + 8;
      P[m][keyBase + lr] = (_Float16)(acc[r] * scale + kbv);
    }
  }
  __syncthreads();
  {
    const int row = lr;
    const int j0 = lh * (S_ / 2), j1 = j0 + (S_ / 2);
    float mx = -1e30f;
    for (int j = j0; j < j1; ++j) mx = fmaxf(mx, (float)P[row][j]);
    mx = fmaxf(mx, __shfl_xor(mx, 16, 32));
    float sum = 0.0f;
    for (int j = j0; j < j1; ++j) {
      float e = __expf((float)P[row][j] - mx);
      sum += e;
      P[row][j] = (_Float16)e;
    }
    sum += __shfl_xor(sum, 16, 32);
    if (lane < 16) invs[lane] = 1.0f / sum;
  }
  __syncthreads();

  v8f o[3];
  for (int i = 0; i < 3; ++i)
    for (int j = 0; j < 8; ++j) o[i][j] = 0.0f;
  for (int kc = 0; kc < S_ / 32; ++kc) {
    v16h ap = *(const v16h*)&P[lr][kc * 32 + lh * 16];
#pragma unroll
    for (int nt = 0; nt < 3; ++nt) {   // only dims 0..47 matter
      const _Float16* Vp = Vt + ((size_t)(b * H_ + h) * HDP_ + nt * 16 + lr) * S_
                              + kc * 32 + lh * 16;
      v16h bv = *(const v16h*)Vp;
      o[nt] = __builtin_amdgcn_wmma_f32_16x16x32_f16(false, ap, false, bv, (short)0, o[nt], false, false);
    }
  }
#pragma unroll
  for (int nt = 0; nt < 3; ++nt) {
    int dim = nt * 16 + lr;
    for (int r = 0; r < 8; ++r) {
      int m = (lane < 16) ? r : r + 8;
      Oout[((size_t)b * S_ + qBase + m) * D_ + h * HD_ + dim] =
          (_Float16)(o[nt][r] * invs[m]);
    }
  }
}

// ---------------------------------------------------------------------------
// Pack q/k/v (f16 mirrors, [B*S, D]) into head-major f16 buffers padded 48->64.
// Qh,Kh: [B,H,S,64]; Vt: [B,H,64,S] (transposed for contiguous B-fragments).
// ---------------------------------------------------------------------------
__global__ __launch_bounds__(64)
void qkv_pack_k(const _Float16* __restrict__ q, const _Float16* __restrict__ k,
                const _Float16* __restrict__ v, _Float16* __restrict__ Qh,
                _Float16* __restrict__ Kh, _Float16* __restrict__ Vt) {
  int tok = blockIdx.x;
  int d = threadIdx.x;
  int b = tok / S_, s = tok % S_;
  for (int h = 0; h < H_; ++h) {
    _Float16 qv = (_Float16)0.f, kv = (_Float16)0.f, vv = (_Float16)0.f;
    if (d < HD_) {
      size_t i = (size_t)tok * D_ + h * HD_ + d;
      qv = q[i]; kv = k[i]; vv = v[i];
    }
    size_t o1 = ((size_t)(b * H_ + h) * S_ + s) * HDP_ + d;
    Qh[o1] = qv;
    Kh[o1] = kv;
    Vt[((size_t)(b * H_ + h) * HDP_ + d) * S_ + s] = vv;
  }
}

// ---------------------------------------------------------------------------
// Input projections: Linear(48/12 -> 384) -> LN -> GELU -> +type (+level)
// ---------------------------------------------------------------------------
__global__ __launch_bounds__(128)
void face_proj_k(const float* __restrict__ ff, const int* __restrict__ bfs,
                 const float* __restrict__ W, const float* __restrict__ bi,
                 const float* __restrict__ g, const float* __restrict__ bb,
                 const float* __restrict__ ftype, const float* __restrict__ lemb,
                 float* __restrict__ F) {
  __shared__ float sin[48];
  __shared__ float sred[128];
  int row = blockIdx.x, t = threadIdx.x;
  if (t < 48) sin[t] = ff[(size_t)row * 48 + t];
  __syncthreads();
  float vals[3]; float ls = 0.f;
  for (int i = 0; i < 3; ++i) {
    int d = t + i * 128;
    float a = bi[d];
    for (int kk = 0; kk < 48; ++kk) a += sin[kk] * W[kk * D_ + d];
    vals[i] = a; ls += a;
  }
  float mean = block_sum(ls, sred) * (1.0f / D_);
  float lv = 0.f;
  for (int i = 0; i < 3; ++i) { float c = vals[i] - mean; lv += c * c; }
  float inv = rsqrtf(block_sum(lv, sred) * (1.0f / D_) + 1e-5f);
  int lvl = bfs[row]; lvl = min(max(lvl, 0), MAXLVL_ - 1);
  for (int i = 0; i < 3; ++i) {
    int d = t + i * 128;
    float x = (vals[i] - mean) * inv * g[d] + bb[d];
    F[(size_t)row * D_ + d] = gelu_f(x) + ftype[d] + lemb[lvl * D_ + d];
  }
}

__global__ __launch_bounds__(128)
void edge_proj_k(const float* __restrict__ ef, const float* __restrict__ W,
                 const float* __restrict__ bi, const float* __restrict__ g,
                 const float* __restrict__ bb, const float* __restrict__ etype,
                 float* __restrict__ E) {
  __shared__ float sin[12];
  __shared__ float sred[128];
  int row = blockIdx.x, t = threadIdx.x;
  if (t < 12) sin[t] = ef[(size_t)row * 12 + t];
  __syncthreads();
  float vals[3]; float ls = 0.f;
  for (int i = 0; i < 3; ++i) {
    int d = t + i * 128;
    float a = bi[d];
    for (int kk = 0; kk < 12; ++kk) a += sin[kk] * W[kk * D_ + d];
    vals[i] = a; ls += a;
  }
  float mean = block_sum(ls, sred) * (1.0f / D_);
  float lv = 0.f;
  for (int i = 0; i < 3; ++i) { float c = vals[i] - mean; lv += c * c; }
  float inv = rsqrtf(block_sum(lv, sred) * (1.0f / D_) + 1e-5f);
  for (int i = 0; i < 3; ++i) {
    int d = t + i * 128;
    float x = (vals[i] - mean) * inv * g[d] + bb[d];
    E[(size_t)row * D_ + d] = gelu_f(x) + etype[d];
  }
}

// ---------------------------------------------------------------------------
// LayerNorm kernels: out32 and/or out16 (either may be null)
// ---------------------------------------------------------------------------
__global__ __launch_bounds__(128)
void ln_k(const float* __restrict__ x, const float* __restrict__ g,
          const float* __restrict__ bb, float* __restrict__ out32,
          _Float16* __restrict__ out16, int nanfix) {
  __shared__ float sred[128];
  int row = blockIdx.x, t = threadIdx.x;
  float vals[3]; float ls = 0.f;
  for (int i = 0; i < 3; ++i) { vals[i] = x[(size_t)row * D_ + t + i * 128]; ls += vals[i]; }
  float mean = block_sum(ls, sred) * (1.0f / D_);
  float lv = 0.f;
  for (int i = 0; i < 3; ++i) { float c = vals[i] - mean; lv += c * c; }
  float inv = rsqrtf(block_sum(lv, sred) * (1.0f / D_) + 1e-5f);
  for (int i = 0; i < 3; ++i) {
    int d = t + i * 128;
    float v = (vals[i] - mean) * inv * g[d] + bb[d];
    if (nanfix) {
      if (!(v == v)) v = 0.0f;
      v = fminf(fmaxf(v, -3.4028234663852886e38f), 3.4028234663852886e38f);
    }
    if (out32) out32[(size_t)row * D_ + d] = v;
    if (out16) out16[(size_t)row * D_ + d] = (_Float16)v;
  }
}

__global__ __launch_bounds__(128)
void gated_ln_k(const float* __restrict__ base, const float* __restrict__ gate,
                const float* __restrict__ msg, const float* __restrict__ rowscale,
                const float* __restrict__ g, const float* __restrict__ bb,
                float* __restrict__ out) {
  __shared__ float sred[128];
  int row = blockIdx.x, t = threadIdx.x;
  float rs = rowscale ? rowscale[row] : 1.0f;
  float vals[3]; float ls = 0.f;
  for (int i = 0; i < 3; ++i) {
    size_t idx = (size_t)row * D_ + t + i * 128;
    vals[i] = base[idx] + gate[idx] * msg[idx] * rs;
    ls += vals[i];
  }
  float mean = block_sum(ls, sred) * (1.0f / D_);
  float lv = 0.f;
  for (int i = 0; i < 3; ++i) { float c = vals[i] - mean; lv += c * c; }
  float inv = rsqrtf(block_sum(lv, sred) * (1.0f / D_) + 1e-5f);
  for (int i = 0; i < 3; ++i) {
    int d = t + i * 128;
    out[(size_t)row * D_ + d] = (vals[i] - mean) * inv * g[d] + bb[d];
  }
}

// ---------------------------------------------------------------------------
// Graph glue kernels (concat kernels emit f16 GEMM operands directly)
// ---------------------------------------------------------------------------
__global__ void valid_k(const unsigned char* __restrict__ em, const int* __restrict__ e2f,
                        float* __restrict__ vm, int* __restrict__ f1, int* __restrict__ f2) {
  int i = blockIdx.x * blockDim.x + threadIdx.x;
  if (i >= B_ * NE_) return;
  int a = e2f[i * 2], c = e2f[i * 2 + 1];
  vm[i] = (em[i] && a >= 0 && c >= 0) ? 1.0f : 0.0f;
  f1[i] = min(max(a, 0), NF_ - 1);
  f2[i] = min(max(c, 0), NF_ - 1);
}

__global__ void fmask_k(const unsigned char* __restrict__ fm, float* __restrict__ out) {
  int i = blockIdx.x * blockDim.x + threadIdx.x;
  if (i < B_ * NF_) out[i] = fm[i] ? 1.0f : 0.0f;
}

__global__ void kbias_k(const unsigned char* __restrict__ fm, const unsigned char* __restrict__ em,
                        float* __restrict__ kb) {
  int i = blockIdx.x * blockDim.x + threadIdx.x;
  if (i >= B_ * S_) return;
  int b = i / S_, s = i % S_;
  unsigned char m = (s < NF_) ? fm[b * NF_ + s] : em[b * NE_ + (s - NF_)];
  kb[i] = m ? 0.0f : -1e9f;
}

__global__ void zero_k(float* __restrict__ p, int n) {
  int i = blockIdx.x * blockDim.x + threadIdx.x;
  if (i < n) p[i] = 0.0f;
}

__global__ void rowmul_k(float* __restrict__ x, const float* __restrict__ vm, int n) {
  int i = blockIdx.x * blockDim.x + threadIdx.x;
  if (i < n) x[i] *= vm[i / D_];
}

__global__ __launch_bounds__(128)
void concat3_k(const float* __restrict__ E, const float* __restrict__ F,
               const int* __restrict__ f1, const int* __restrict__ f2,
               _Float16* __restrict__ dst) {
  int e = blockIdx.x, t = threadIdx.x;
  int b = e / NE_;
  size_t r1 = (size_t)(b * NF_ + f1[e]) * D_;
  size_t r2 = (size_t)(b * NF_ + f2[e]) * D_;
  for (int i = 0; i < 3; ++i) {
    int d = t + i * 128;
    dst[(size_t)e * (3 * D_) + d]          = (_Float16)E[(size_t)e * D_ + d];
    dst[(size_t)e * (3 * D_) + D_ + d]     = (_Float16)F[r1 + d];
    dst[(size_t)e * (3 * D_) + 2 * D_ + d] = (_Float16)F[r2 + d];
  }
}

__global__ __launch_bounds__(128)
void concat2_k(_Float16* __restrict__ dst, const float* __restrict__ a,
               const float* __restrict__ bsrc) {
  int row = blockIdx.x, t = threadIdx.x;
  for (int i = 0; i < 3; ++i) {
    int d = t + i * 128;
    dst[(size_t)row * (2 * D_) + d]      = (_Float16)a[(size_t)row * D_ + d];
    dst[(size_t)row * (2 * D_) + D_ + d] = (_Float16)bsrc[(size_t)row * D_ + d];
  }
}

__global__ __launch_bounds__(128)
void scatter_k(const float* __restrict__ E, const float* __restrict__ vm,
               const int* __restrict__ f1, const int* __restrict__ f2,
               float* __restrict__ fmsg, float* __restrict__ fcnt) {
  int e = blockIdx.x, t = threadIdx.x;
  float v = vm[e];
  if (v == 0.0f) return;   // uniform per block
  int b = e / NE_;
  size_t r1 = (size_t)(b * NF_ + f1[e]) * D_;
  size_t r2 = (size_t)(b * NF_ + f2[e]) * D_;
  for (int i = 0; i < 3; ++i) {
    int d = t + i * 128;
    float c = E[(size_t)e * D_ + d] * v;
    atomicAdd(&fmsg[r1 + d], c);
    atomicAdd(&fmsg[r2 + d], c);
  }
  if (t == 0) {
    atomicAdd(&fcnt[b * NF_ + f1[e]], v);
    atomicAdd(&fcnt[b * NF_ + f2[e]], v);
  }
}

__global__ __launch_bounds__(128)
void face_cat_k(const float* __restrict__ F, const float* __restrict__ fmsg,
                const float* __restrict__ fcnt, _Float16* __restrict__ dst) {
  int row = blockIdx.x, t = threadIdx.x;
  float inv = 1.0f / (fcnt[row] + 1e-8f);
  for (int i = 0; i < 3; ++i) {
    int d = t + i * 128;
    dst[(size_t)row * (2 * D_) + d]      = (_Float16)F[(size_t)row * D_ + d];
    dst[(size_t)row * (2 * D_) + D_ + d] = (_Float16)(fmsg[(size_t)row * D_ + d] * inv);
  }
}

__global__ void buildX_k(const float* __restrict__ F, const float* __restrict__ E,
                         float* __restrict__ X) {
  size_t i = (size_t)blockIdx.x * 256 + threadIdx.x;
  if (i >= (size_t)B_ * S_ * D_) return;
  int d = (int)(i % D_);
  size_t t2 = i / D_;
  int s = (int)(t2 % S_);
  int b = (int)(t2 / S_);
  X[i] = (s < NF_) ? F[((size_t)b * NF_ + s) * D_ + d]
                   : E[((size_t)b * NE_ + (s - NF_)) * D_ + d];
}

__global__ void maskout_k(const unsigned char* __restrict__ fm,
                          const unsigned char* __restrict__ em,
                          float* __restrict__ out) {
  int i = blockIdx.x * blockDim.x + threadIdx.x;
  if (i >= B_ * S_) return;
  int b = i / S_, s = i % S_;
  unsigned char m = (s < NF_) ? fm[b * NF_ + s] : em[b * NE_ + (s - NF_)];
  out[i] = m ? 1.0f : 0.0f;
}

// ---------------------------------------------------------------------------
// Orchestration
// ---------------------------------------------------------------------------
extern "C" void kernel_launch(void* const* d_in, const int* in_sizes, int n_in,
                              void* d_out, int out_size, void* d_ws, size_t ws_size,
                              hipStream_t stream) {
  (void)in_sizes; (void)n_in; (void)out_size; (void)ws_size;

  const float* face_feats = (const float*)d_in[0];
  const float* edge_feats = (const float*)d_in[1];
  const unsigned char* face_mask = (const unsigned char*)d_in[2];
  const unsigned char* edge_mask = (const unsigned char*)d_in[3];
  const int* e2f = (const int*)d_in[4];
  const int* bfs = (const int*)d_in[5];
  auto Pp = [&](int i) { return (const float*)d_in[i]; };
  const int MSG0 = 17, TF0 = 65;  // flattened param dict offsets

  float* ws = (float*)d_ws;
  // ---- persistent f32 buffers ----
  size_t oF  = 0;
  size_t oE  = oF + (size_t)B_ * NF_ * D_;
  size_t oX  = oE + (size_t)B_ * NE_ * D_;
  size_t oKB = oX + (size_t)B_ * S_ * D_;
  size_t oVM = oKB + B_ * S_;
  size_t oFM = oVM + B_ * NE_;
  size_t oI1 = oFM + B_ * NF_;
  size_t oI2 = oI1 + B_ * NE_;
  size_t oWA = oI2 + B_ * NE_;          // f16 weight arena (9289728 floats)
  size_t oSC = oWA + 9289728;           // phase-overlapped scratch

  float* Fb = ws + oF;
  float* Eb = ws + oE;
  float* Xb = ws + oX;
  float* KB = ws + oKB;
  float* VM = ws + oVM;
  float* FM = ws + oFM;
  int* I1 = (int*)(ws + oI1);
  int* I2 = (int*)(ws + oI2);

  // ---- weight arena: transpose+convert every GEMM weight to f16 [N][K] ----
  _Float16* warena = (_Float16*)(ws + oWA);
  _Float16* WT16[163] = {};
  size_t wofs = 0;
  auto cvt = [&](int idx, int Kd, int Nd) {
    WT16[idx] = warena + wofs;
    wofs += (size_t)Kd * Nd;
    int tot = Kd * Nd;
    wT16_k<<<(tot + 255) / 256, 256, 0, stream>>>(Pp(idx), WT16[idx], Kd, Nd);
  };
  for (int l = 0; l < 3; ++l) {
    const int pb = MSG0 + 16 * l;
    cvt(pb + 0, 1152, 768);  cvt(pb + 2, 768, 384);
    cvt(pb + 4, 768, 768);   cvt(pb + 6, 768, 384);
    cvt(pb + 8, 768, 384);   cvt(pb + 10, 768, 384);
  }
  for (int t = 0; t < 6; ++t) {
    const int pb = TF0 + 16 * t;
    cvt(pb + 0, 384, 384); cvt(pb + 1, 384, 384);
    cvt(pb + 2, 384, 384); cvt(pb + 3, 384, 384);
    cvt(pb + 8, 384, 1536); cvt(pb + 10, 1536, 384);
  }

  // ---- msg-phase scratch ----
  float* msgE  = ws + oSC;                  // 2048x384 f32
  float* gateE = msgE + (size_t)2048 * 384;
  float* msgF  = gateE + (size_t)2048 * 384;
  float* gateF = msgF + (size_t)1024 * 384;
  float* fmsg  = gateF + (size_t)1024 * 384;
  float* fcnt  = fmsg + (size_t)1024 * 384;
  _Float16* cat3_16  = (_Float16*)(fcnt + 1024);
  _Float16* hidE16   = cat3_16 + (size_t)2048 * 1152;
  _Float16* cat2E16  = hidE16 + (size_t)2048 * 768;
  _Float16* catF16   = cat2E16 + (size_t)2048 * 768;
  _Float16* hidF16   = catF16 + (size_t)1024 * 768;
  _Float16* cat2F16  = hidF16 + (size_t)1024 * 768;

  // ---- tf-phase scratch (reuses msg region; phases are sequential) ----
  _Float16* y16   = (_Float16*)(ws + oSC);
  _Float16* q16   = y16 + (size_t)3072 * 384;
  _Float16* k16   = q16 + (size_t)3072 * 384;
  _Float16* v16p  = k16 + (size_t)3072 * 384;
  _Float16* o16   = v16p + (size_t)3072 * 384;
  _Float16* hid16 = o16 + (size_t)3072 * 384;
  _Float16* Qh = hid16 + (size_t)3072 * 1536;
  _Float16* Kh = Qh + (size_t)B_ * H_ * S_ * HDP_;
  _Float16* Vt = Kh + (size_t)B_ * H_ * S_ * HDP_;

  // ---- masks / indices / projections ----
  valid_k<<<(B_ * NE_ + 127) / 128, 128, 0, stream>>>(edge_mask, e2f, VM, I1, I2);
  fmask_k<<<(B_ * NF_ + 127) / 128, 128, 0, stream>>>(face_mask, FM);
  kbias_k<<<(B_ * S_ + 127) / 128, 128, 0, stream>>>(face_mask, edge_mask, KB);
  face_proj_k<<<B_ * NF_, 128, 0, stream>>>(face_feats, bfs, Pp(6), Pp(7), Pp(8), Pp(9),
                                            Pp(14), Pp(16), Fb);
  edge_proj_k<<<B_ * NE_, 128, 0, stream>>>(edge_feats, Pp(10), Pp(11), Pp(12), Pp(13),
                                            Pp(15), Eb);

  // ---- message-passing layers ----
  for (int l = 0; l < 3; ++l) {
    const int pb = MSG0 + 16 * l;
    concat3_k<<<B_ * NE_, 128, 0, stream>>>(Eb, Fb, I1, I2, cat3_16);
    gemm16_k<1><<<dim3(6, 32), 128, 0, stream>>>(cat3_16, WT16[pb + 0], Pp(pb + 1),
                                                 nullptr, nullptr, hidE16, 2048, 768, 1152);
    gemm16_k<0><<<dim3(3, 32), 128, 0, stream>>>(hidE16, WT16[pb + 2], Pp(pb + 3),
                                                 nullptr, msgE, nullptr, 2048, 384, 768);
    rowmul_k<<<(2048 * 384 + 255) / 256, 256, 0, stream>>>(msgE, VM, 2048 * 384);
    concat2_k<<<2048, 128, 0, stream>>>(cat2E16, Eb, msgE);
    gemm16_k<2><<<dim3(3, 32), 128, 0, stream>>>(cat2E16, WT16[pb + 8], Pp(pb + 9),
                                                 nullptr, gateE, nullptr, 2048, 384, 768);
    gated_ln_k<<<2048, 128, 0, stream>>>(Eb, gateE, msgE, nullptr, Pp(pb + 12), Pp(pb + 13), Eb);
    zero_k<<<(1024 * 384 + 1024 + 255) / 256, 256, 0, stream>>>(fmsg, 1024 * 384 + 1024);
    scatter_k<<<B_ * NE_, 128, 0, stream>>>(Eb, VM, I1, I2, fmsg, fcnt);
    face_cat_k<<<B_ * NF_, 128, 0, stream>>>(Fb, fmsg, fcnt, catF16);
    gemm16_k<1><<<dim3(6, 16), 128, 0, stream>>>(catF16, WT16[pb + 4], Pp(pb + 5),
                                                 nullptr, nullptr, hidF16, 1024, 768, 768);
    gemm16_k<0><<<dim3(3, 16), 128, 0, stream>>>(hidF16, WT16[pb + 6], Pp(pb + 7),
                                                 nullptr, msgF, nullptr, 1024, 384, 768);
    concat2_k<<<1024, 128, 0, stream>>>(cat2F16, Fb, msgF);
    gemm16_k<2><<<dim3(3, 16), 128, 0, stream>>>(cat2F16, WT16[pb + 10], Pp(pb + 11),
                                                 nullptr, gateF, nullptr, 1024, 384, 768);
    gated_ln_k<<<1024, 128, 0, stream>>>(Fb, gateF, msgF, FM, Pp(pb + 14), Pp(pb + 15), Fb);
  }

  // ---- assemble X = [F ; E] ----
  buildX_k<<<(int)(((size_t)B_ * S_ * D_ + 255) / 256), 256, 0, stream>>>(Fb, Eb, Xb);

  // ---- transformer layers ----
  for (int t = 0; t < 6; ++t) {
    const int pb = TF0 + 16 * t;
    ln_k<<<3072, 128, 0, stream>>>(Xb, Pp(pb + 12), Pp(pb + 13), nullptr, y16, 0);
    gemm16_k<0><<<dim3(3, 48), 128, 0, stream>>>(y16, WT16[pb + 0], Pp(pb + 4),
                                                 nullptr, nullptr, q16, 3072, 384, 384);
    gemm16_k<0><<<dim3(3, 48), 128, 0, stream>>>(y16, WT16[pb + 1], Pp(pb + 5),
                                                 nullptr, nullptr, k16, 3072, 384, 384);
    gemm16_k<0><<<dim3(3, 48), 128, 0, stream>>>(y16, WT16[pb + 2], Pp(pb + 6),
                                                 nullptr, nullptr, v16p, 3072, 384, 384);
    qkv_pack_k<<<B_ * S_, 64, 0, stream>>>(q16, k16, v16p, Qh, Kh, Vt);
    attn_k<<<dim3(S_ / 16, H_, B_), 32, 0, stream>>>(Qh, Kh, Vt, KB, o16);
    gemm16_k<0><<<dim3(3, 48), 128, 0, stream>>>(o16, WT16[pb + 3], Pp(pb + 7),
                                                 Xb, Xb, nullptr, 3072, 384, 384);
    ln_k<<<3072, 128, 0, stream>>>(Xb, Pp(pb + 14), Pp(pb + 15), nullptr, y16, 0);
    gemm16_k<1><<<dim3(12, 48), 128, 0, stream>>>(y16, WT16[pb + 8], Pp(pb + 9),
                                                  nullptr, nullptr, hid16, 3072, 1536, 384);
    gemm16_k<0><<<dim3(3, 48), 128, 0, stream>>>(hid16, WT16[pb + 10], Pp(pb + 11),
                                                 Xb, Xb, nullptr, 3072, 384, 1536);
  }

  // ---- final LN (nan_to_num) + mask output ----
  ln_k<<<3072, 128, 0, stream>>>(Xb, Pp(161), Pp(162), (float*)d_out, nullptr, 1);
  maskout_k<<<(B_ * S_ + 255) / 256, 256, 0, stream>>>(face_mask, edge_mask,
                                                       (float*)d_out + (size_t)B_ * S_ * D_);
}
